// TrafficXLSTM_52072183497479
// MI455X (gfx1250) — matrix operand order repaired and checked by reading, not compiled, not run on hardware
//
#include <hip/hip_runtime.h>
#include <hip/hip_bf16.h>

// ---------------------------------------------------------------------------
// TrafficXLSTM forward for MI455X (gfx1250, wave32, WMMA bf16 16x16x32)
// ---------------------------------------------------------------------------

typedef __attribute__((ext_vector_type(16))) __bf16 v16bf;
typedef __attribute__((ext_vector_type(8)))  float  v8f;

#define NTOK 16384   // B*S = 64*256

static __device__ __forceinline__ unsigned f32_to_bf16(float f) {
  unsigned u = __builtin_bit_cast(unsigned, f);
  unsigned r = u + 0x7FFFu + ((u >> 16) & 1u);   // round-to-nearest-even
  return r >> 16;
}

// ---------------------------------------------------------------------------
// Generic strided/batched GEMM: C = alpha * A@B (+ res), fp32 in/out,
// bf16 WMMA with fp32 accumulate.
//   workgroup  = 256 threads = 8 wave32s arranged 4(M) x 2(N)
//   block tile = 64(M) x 64(N), K step 64
//   each wave  = two 16x16 accumulators -> 4 v_wmma per K iteration
// LDS tiles are stored FRAGMENT-MAJOR as packed bf16 pairs, so each lane's
// WMMA operand is one contiguous 32B vector load (ds_load_b128 pairs).
// Batch index z decomposed as zo*inner + zi for head-sliced attention batches.
// Requires M%64==0, N%64==0, K%64==0 (true for every call site here).
// ---------------------------------------------------------------------------
__global__ __launch_bounds__(256)
void k_gemm_bf16(const float* __restrict__ A, const float* __restrict__ B,
                 float* __restrict__ C, const float* __restrict__ res,
                 int M, int N, int Kd,
                 long long sAm, long long sAk, long long sBk, long long sBn,
                 long long sCm, int innerB,
                 long long oA, long long iA, long long oB, long long iB,
                 long long oC, long long iC, float alpha)
{
  __shared__ __attribute__((aligned(32))) unsigned int ALds[2048];  // 8 KB
  __shared__ __attribute__((aligned(32))) unsigned int BLds[2048];  // 8 KB

  const int z  = blockIdx.z;
  const int zo = z / innerB, zi = z % innerB;
  A += (long long)zo * oA + (long long)zi * iA;
  B += (long long)zo * oB + (long long)zi * iB;
  const long long cOff = (long long)zo * oC + (long long)zi * iC;
  C += cOff;
  if (res) res += cOff;

  const int tid  = threadIdx.x;
  const int lane = tid & 31;
  const int wave = tid >> 5;
  const int wm   = wave & 3;     // 4 M sub-tiles
  const int wn   = wave >> 2;    // 2 N sub-tiles (x2 accumulators each)
  const int half = lane >> 4;
  const int l16  = lane & 15;

  const int mBase = blockIdx.y * 64;
  const int nBase = blockIdx.x * 64;

  v8f acc0 = {0.f, 0.f, 0.f, 0.f, 0.f, 0.f, 0.f, 0.f};
  v8f acc1 = {0.f, 0.f, 0.f, 0.f, 0.f, 0.f, 0.f, 0.f};

  for (int k0 = 0; k0 < Kd; k0 += 64) {
    // ---- stage A (64 rows x 64 K), fragment-major packed bf16 pairs ----
    // word w -> (am, ks, ln, i2):  fragment element i = i2*2 of lane ln in
    // A sub-tile am, K half ks.  Element i maps to K = half*8+i (i<8) or
    // 16+half*8+(i-8), so a word always covers two consecutive K columns.
#pragma unroll
    for (int p = 0; p < 8; ++p) {
      int w  = p * 256 + tid;
      int i  = (w & 7) * 2;
      int ln = (w >> 3) & 31;
      int ks = (w >> 8) & 1;
      int am = w >> 9;
      int hf = ln >> 4;
      int row = mBase + am * 16 + (ln & 15);
      int kk  = (i < 8) ? (hf * 8 + i) : (16 + hf * 8 + (i - 8));
      long long gb = (long long)row * sAm + (long long)(k0 + ks * 32 + kk) * sAk;
      unsigned lo = f32_to_bf16(A[gb]);
      unsigned hi = f32_to_bf16(A[gb + sAk]);
      ALds[w] = lo | (hi << 16);
    }
    // ---- stage B (64 K x 64 N), fragment-major ----
    // element i of lane ln maps to K = half*8+i (i<8) or 16+half*8+(i-8)  (always pair-consecutive)
#pragma unroll
    for (int p = 0; p < 8; ++p) {
      int w  = p * 256 + tid;
      int i  = (w & 7) * 2;
      int ln = (w >> 3) & 31;
      int ks = (w >> 8) & 1;
      int ns = (w >> 9) & 1;
      int bn = w >> 10;
      int hf = ln >> 4;
      int col = nBase + bn * 32 + ns * 16 + (ln & 15);
      int kk  = (i < 8) ? (hf * 8 + i) : (16 + hf * 8 + (i - 8));
      long long gb = (long long)(k0 + ks * 32 + kk) * sBk + (long long)col * sBn;
      unsigned lo = f32_to_bf16(B[gb]);
      unsigned hi = f32_to_bf16(B[gb + sBk]);
      BLds[w] = lo | (hi << 16);
    }
    if (k0 + 64 < Kd)                      // hint next K slab toward caches
      __builtin_prefetch(&A[(long long)(mBase + (tid >> 2)) * sAm +
                            (long long)(k0 + 64) * sAk], 0, 1);
    __syncthreads();

    // one contiguous 32B vector load per fragment
    v16bf a0  = *(const v16bf*)&ALds[((wm * 2 + 0) * 32 + lane) * 8];
    v16bf a1  = *(const v16bf*)&ALds[((wm * 2 + 1) * 32 + lane) * 8];
    v16bf b00 = *(const v16bf*)&BLds[(((wn * 2 + 0) * 2 + 0) * 32 + lane) * 8];
    v16bf b01 = *(const v16bf*)&BLds[(((wn * 2 + 0) * 2 + 1) * 32 + lane) * 8];
    v16bf b10 = *(const v16bf*)&BLds[(((wn * 2 + 1) * 2 + 0) * 32 + lane) * 8];
    v16bf b11 = *(const v16bf*)&BLds[(((wn * 2 + 1) * 2 + 1) * 32 + lane) * 8];

    acc0 = __builtin_amdgcn_wmma_f32_16x16x32_bf16(false, a0, false, b00,
                                                   (short)0, acc0, false, false);
    acc0 = __builtin_amdgcn_wmma_f32_16x16x32_bf16(false, a1, false, b01,
                                                   (short)0, acc0, false, false);
    acc1 = __builtin_amdgcn_wmma_f32_16x16x32_bf16(false, a0, false, b10,
                                                   (short)0, acc1, false, false);
    acc1 = __builtin_amdgcn_wmma_f32_16x16x32_bf16(false, a1, false, b11,
                                                   (short)0, acc1, false, false);
    __syncthreads();
  }

#pragma unroll
  for (int r = 0; r < 8; ++r) {            // C layout: VGPR r -> M = half*8+r
    int row = mBase + wm * 16 + half * 8 + r;
    int c0  = nBase + wn * 32 + l16;
    int c1  = c0 + 16;
    float v0 = alpha * acc0[r];
    float v1 = alpha * acc1[r];
    if (res) {
      v0 += res[(long long)row * sCm + c0];
      v1 += res[(long long)row * sCm + c1];
    }
    C[(long long)row * sCm + c0] = v0;
    C[(long long)row * sCm + c1] = v1;
  }
}

// ---------------------------------------------------------------------------
// Elementwise / reduction kernels (wave32 shuffles, no wave64 assumptions)
// ---------------------------------------------------------------------------
__global__ void k_embed(const float* __restrict__ x, const float* __restrict__ W,
                        const float* __restrict__ b, float* __restrict__ h) {
  long long i = (long long)blockIdx.x * blockDim.x + threadIdx.x;
  if (i >= (long long)NTOK * 128) return;
  long long t = i >> 7; int d = i & 127;
  float a = b[d];
#pragma unroll
  for (int j = 0; j < 3; ++j) a += x[t * 3 + j] * W[j * 128 + d];
  h[i] = a;
}

__global__ __launch_bounds__(128)
void k_ln(const float* __restrict__ x, float* __restrict__ y,
          const float* __restrict__ w) {
  int wave = threadIdx.x >> 5, lane = threadIdx.x & 31;
  long long t = (long long)blockIdx.x * 4 + wave;          // grid = NTOK/4
  float v[4]; float s = 0.f, s2 = 0.f;
#pragma unroll
  for (int i = 0; i < 4; ++i) {
    v[i] = x[t * 128 + lane + 32 * i]; s += v[i]; s2 += v[i] * v[i];
  }
  for (int off = 16; off; off >>= 1) {
    s += __shfl_xor(s, off, 32); s2 += __shfl_xor(s2, off, 32);
  }
  float mu = s * (1.f / 128.f), var = s2 * (1.f / 128.f) - mu * mu;
  float r = rsqrtf(var + 1e-6f);
#pragma unroll
  for (int i = 0; i < 4; ++i) {
    int d = lane + 32 * i;
    y[t * 128 + d] = (v[i] - mu) * r * w[d];
  }
}

__global__ void k_conv_silu(const float* __restrict__ src, int rowStride, int Cc,
                            const float* __restrict__ w, const float* __restrict__ bias,
                            float* __restrict__ dst) {
  long long i = (long long)blockIdx.x * blockDim.x + threadIdx.x;
  if (i >= (long long)NTOK * Cc) return;
  long long t = i / Cc; int c = (int)(i % Cc);
  int s = (int)(t & 255);
  float a = bias[c];
#pragma unroll
  for (int k = 0; k < 4; ++k) {
    int sp = s - 3 + k;
    if (sp >= 0) a += src[(t - 3 + k) * rowStride + c] * w[c * 4 + k];
  }
  dst[t * (long long)Cc + c] = a / (1.f + expf(-a));
}

__global__ void k_headwise(const float* __restrict__ xc, const float* __restrict__ xm,
                           int xmStride, const float* __restrict__ Wq,
                           const float* __restrict__ Wk, const float* __restrict__ Wv,
                           float* __restrict__ q, float* __restrict__ k,
                           float* __restrict__ v) {
  long long i = (long long)blockIdx.x * blockDim.x + threadIdx.x;
  if (i >= (long long)NTOK * 64) return;
  long long t = i >> 6; int nb = (int)(i & 63);
  float xa[4], xb[4];
#pragma unroll
  for (int j = 0; j < 4; ++j) {
    xa[j] = xc[t * 256 + nb * 4 + j];
    xb[j] = xm[t * xmStride + nb * 4 + j];
  }
#pragma unroll
  for (int l = 0; l < 4; ++l) {
    float aq = 0.f, ak = 0.f, av = 0.f;
#pragma unroll
    for (int j = 0; j < 4; ++j) {
      aq += xa[j] * Wq[nb * 16 + j * 4 + l];
      ak += xa[j] * Wk[nb * 16 + j * 4 + l];
      av += xb[j] * Wv[nb * 16 + j * 4 + l];
    }
    q[t * 256 + nb * 4 + l] = aq;
    k[t * 256 + nb * 4 + l] = ak;
    v[t * 256 + nb * 4 + l] = av;
  }
}

__global__ void k_gates(const float* __restrict__ q, const float* __restrict__ k,
                        const float* __restrict__ v, const float* __restrict__ Wig,
                        const float* __restrict__ big, const float* __restrict__ Wfg,
                        const float* __restrict__ bfg, float* __restrict__ igT,
                        float* __restrict__ fgT) {
  long long i = (long long)blockIdx.x * blockDim.x + threadIdx.x;
  if (i >= (long long)NTOK * 4) return;
  long long t = i >> 2; int h = (int)(i & 3);
  float ai = big[h], af = bfg[h];
  for (int f = 0; f < 256; ++f) {
    float qv = q[t * 256 + f], kv = k[t * 256 + f], vv = v[t * 256 + f];
    ai += qv * Wig[f * 4 + h] + kv * Wig[(256 + f) * 4 + h] + vv * Wig[(512 + f) * 4 + h];
    af += qv * Wfg[f * 4 + h] + kv * Wfg[(256 + f) * 4 + h] + vv * Wfg[(512 + f) * 4 + h];
  }
  long long b = t >> 8, s = t & 255;
  long long o = (b * 4 + h) * 256 + s;                  // [b,h,s]
  igT[o] = ai; fgT[o] = af;
}

__global__ __launch_bounds__(256)
void k_cumsum_logsig(const float* __restrict__ fgT, float* __restrict__ cs) {
  __shared__ float sc[256];
  long long bh = blockIdx.x; int s = threadIdx.x;
  float x = fgT[bh * 256 + s];
  float ls = fminf(x, 0.f) - log1pf(expf(-fabsf(x)));   // log_sigmoid
  sc[s] = ls; __syncthreads();
  for (int off = 1; off < 256; off <<= 1) {
    float t = (s >= off) ? sc[s - off] : 0.f;
    __syncthreads();
    sc[s] += t; __syncthreads();
  }
  cs[bh * 256 + s] = sc[s];
}

__global__ __launch_bounds__(256)
void k_rownorm(float* __restrict__ sc, const float* __restrict__ cs,
               const float* __restrict__ igT) {
  __shared__ float red[256];
  long long z = blockIdx.x;            // bh*S + i
  int i = (int)(z & 255); long long bh = z >> 8;
  int j = threadIdx.x;
  long long base = bh * 65536 + (long long)i * 256;
  float csi = cs[bh * 256 + i];
  float lD = (j <= i) ? (csi - cs[bh * 256 + j] + igT[bh * 256 + j]) : -1e30f;
  red[j] = lD; __syncthreads();
  for (int off = 128; off; off >>= 1) {
    if (j < off) red[j] = fmaxf(red[j], red[j + off]);
    __syncthreads();
  }
  float mx = red[0]; __syncthreads();
  float Cv = sc[base + j] * expf(lD - mx);
  red[j] = Cv; __syncthreads();
  for (int off = 128; off; off >>= 1) {
    if (j < off) red[j] += red[j + off];
    __syncthreads();
  }
  float n = fmaxf(fabsf(red[0]), expf(-mx)) + 1e-6f;
  sc[base + j] = Cv / n;
}

__global__ __launch_bounds__(128)
void k_mhnorm(const float* __restrict__ src, float* __restrict__ dst,
              const float* __restrict__ w, const float* __restrict__ addTo,
              int nHead, int hd) {
  int wave = threadIdx.x >> 5, lane = threadIdx.x & 31;
  long long u = (long long)blockIdx.x * 4 + wave;      // token*nHead + head
  long long t = u / nHead; int h = (int)(u % nHead);
  int E = hd >> 5;                                     // 1 (hd=32) or 2 (hd=64)
  int Dfull = nHead * hd;
  float v[2] = {0.f, 0.f}; float s = 0.f, s2 = 0.f;
  for (int e = 0; e < E; ++e) {
    v[e] = src[t * Dfull + h * hd + lane + 32 * e];
    s += v[e]; s2 += v[e] * v[e];
  }
  for (int off = 16; off; off >>= 1) {
    s += __shfl_xor(s, off, 32); s2 += __shfl_xor(s2, off, 32);
  }
  float inv = 1.f / (float)hd;
  float mu = s * inv, var = s2 * inv - mu * mu;
  float r = rsqrtf(var + 1e-6f);
  for (int e = 0; e < E; ++e) {
    int d = h * hd + lane + 32 * e;
    float o = (v[e] - mu) * r * w[d];
    if (addTo) o += addTo[t * Dfull + d];
    dst[t * Dfull + d] = o;
  }
}

__global__ void k_mtail(const float* __restrict__ hn, const float* __restrict__ xc,
                        const float* __restrict__ up, const float* __restrict__ skip,
                        float* __restrict__ g) {
  long long i = (long long)blockIdx.x * blockDim.x + threadIdx.x;
  if (i >= (long long)NTOK * 256) return;
  long long t = i >> 8; int c = (int)(i & 255);
  float z = up[t * 512 + 256 + c];
  float sz = z / (1.f + expf(-z));
  g[i] = (hn[i] + skip[c] * xc[i]) * sz;
}

__global__ void k_sgates(const float* __restrict__ xc, const float* __restrict__ x,
                         const float* __restrict__ Wg, const float* __restrict__ bg,
                         float* __restrict__ g0, float* __restrict__ g1,
                         float* __restrict__ g2, float* __restrict__ g3) {
  long long i = (long long)blockIdx.x * blockDim.x + threadIdx.x;
  if (i >= (long long)NTOK * 128) return;
  long long t = i >> 7; int d = (int)(i & 127);
  int n = d >> 5, l = d & 31;
  float a0 = bg[0 * 128 + d], a1 = bg[1 * 128 + d];
  float a2 = bg[2 * 128 + d], a3 = bg[3 * 128 + d];
  for (int kk = 0; kk < 32; ++kk) {
    float xcv = xc[t * 128 + n * 32 + kk];
    float xv  = x [t * 128 + n * 32 + kk];
    int wb = n * 1024 + kk * 32 + l;
    a0 += xcv * Wg[0 * 4096 + wb];
    a1 += xcv * Wg[1 * 4096 + wb];
    a2 += xv  * Wg[2 * 4096 + wb];
    a3 += xv  * Wg[3 * 4096 + wb];
  }
  g0[i] = a0; g1[i] = a1; g2[i] = a2; g3[i] = a3;
}

// sLSTM recurrence: one wave per (b, head); R resident in LDS.
__global__ __launch_bounds__(32)
void k_scan(const float* __restrict__ g0, const float* __restrict__ g1,
            const float* __restrict__ g2, const float* __restrict__ g3,
            const float* __restrict__ R, float* __restrict__ ys) {
  __shared__ float Rl[4 * 32 * 32];
  __shared__ float hsh[32];
  int b = blockIdx.x >> 2, n = blockIdx.x & 3, l = threadIdx.x;
  for (int idx = l; idx < 4096; idx += 32) {
    int g = idx >> 10, rem = idx & 1023;
    Rl[idx] = R[(long long)g * 4096 + n * 1024 + rem];
  }
  hsh[l] = 0.f;
  float c = 0.f, nn = 0.f, m = 0.f;
  __syncthreads();
  for (int s = 0; s < 256; ++s) {
    float r0 = 0.f, r1 = 0.f, r2 = 0.f, r3 = 0.f;
#pragma unroll 4
    for (int kk = 0; kk < 32; ++kk) {
      float hv = hsh[kk];
      r0 += hv * Rl[0 * 1024 + kk * 32 + l];
      r1 += hv * Rl[1 * 1024 + kk * 32 + l];
      r2 += hv * Rl[2 * 1024 + kk * 32 + l];
      r3 += hv * Rl[3 * 1024 + kk * 32 + l];
    }
    long long gi = ((long long)b * 256 + s) * 128 + n * 32 + l;
    float ir = g0[gi] + r0, fr = g1[gi] + r1;
    float zr = g2[gi] + r2, orr = g3[gi] + r3;
    float mnew = fmaxf(fr + m, ir);
    float iv = expf(ir - mnew), fv = expf(fr + m - mnew);
    float zt = tanhf(zr);
    float ov = 1.f / (1.f + expf(-orr));
    c  = fv * c + iv * zt;
    nn = fv * nn + iv;
    float hv = ov * c / (nn + 1e-8f);
    m = mnew;
    ys[gi] = hv;
    __syncthreads();
    hsh[l] = hv;
    __syncthreads();
  }
}

__global__ void k_geglu(const float* __restrict__ u, float* __restrict__ o) {
  long long i = (long long)blockIdx.x * blockDim.x + threadIdx.x;
  if (i >= (long long)NTOK * 192) return;
  long long t = i / 192; int c = (int)(i % 192);
  float g = u[t * 384 + c], vv = u[t * 384 + 192 + c];
  float gg = 0.5f * g * (1.f + tanhf(0.7978845608f * (g + 0.044715f * g * g * g)));
  o[i] = gg * vv;
}

__global__ __launch_bounds__(128)
void k_final(const float* __restrict__ h, const float* __restrict__ w,
             const float* __restrict__ Wout, const float* __restrict__ bout,
             float* __restrict__ out) {
  int wave = threadIdx.x >> 5, lane = threadIdx.x & 31;
  int b = blockIdx.x * 4 + wave;                       // grid = 16 -> 64 rows
  long long base = ((long long)b * 256 + 255) * 128;   // last token
  float v[4]; float s = 0.f, s2 = 0.f;
#pragma unroll
  for (int i = 0; i < 4; ++i) {
    v[i] = h[base + lane + 32 * i]; s += v[i]; s2 += v[i] * v[i];
  }
  for (int off = 16; off; off >>= 1) {
    s += __shfl_xor(s, off, 32); s2 += __shfl_xor(s2, off, 32);
  }
  float mu = s * (1.f / 128.f), var = s2 * (1.f / 128.f) - mu * mu;
  float r = rsqrtf(var + 1e-6f);
  float xn[4];
#pragma unroll
  for (int i = 0; i < 4; ++i) {
    int d = lane + 32 * i;
    xn[i] = (v[i] - mu) * r * w[d];
  }
#pragma unroll
  for (int j = 0; j < 3; ++j) {
    float a = 0.f;
#pragma unroll
    for (int i = 0; i < 4; ++i) a += xn[i] * Wout[(lane + 32 * i) * 3 + j];
    for (int off = 16; off; off >>= 1) a += __shfl_xor(a, off, 32);
    if (lane == 0) out[b * 3 + j] = a + bout[j];
  }
}

// ---------------------------------------------------------------------------
// Workspace layout (floats).  Big `scores` region is reused for hn/gemmin/FFN.
// ---------------------------------------------------------------------------
static const long long OFF_H   = 0;
static const long long OFF_LN  = OFF_H   + (long long)NTOK * 128;
static const long long OFF_UP  = OFF_LN  + (long long)NTOK * 128;
static const long long OFF_XC  = OFF_UP  + (long long)NTOK * 512;
static const long long OFF_Q   = OFF_XC  + (long long)NTOK * 256;
static const long long OFF_K   = OFF_Q   + (long long)NTOK * 256;
static const long long OFF_V   = OFF_K   + (long long)NTOK * 256;
static const long long OFF_IG  = OFF_V   + (long long)NTOK * 256;
static const long long OFF_FG  = OFF_IG  + 65536;
static const long long OFF_CS  = OFF_FG  + 65536;
static const long long OFF_HAT = OFF_CS  + 65536;
static const long long OFF_SC  = OFF_HAT + (long long)NTOK * 256;
// aliases into the scores region (valid when scores dead):
static const long long OFF_HN  = OFF_SC;
static const long long OFF_GIN = OFF_SC + (long long)NTOK * 256;
static const long long OFF_FFU = OFF_SC;
static const long long OFF_FFG = OFF_SC + (long long)NTOK * 384;
// sLSTM aliases into q/k/v regions:
static const long long OFF_G0  = OFF_Q;                            // 4x NTOK*128
static const long long OFF_YS  = OFF_V;

static inline void gemm(hipStream_t st, const float* A, const float* B, float* C,
                        const float* res, int M, int N, int K,
                        long long sAm, long long sAk, long long sBk, long long sBn,
                        long long sCm, int batch, int innerB,
                        long long oA, long long iA, long long oB, long long iB,
                        long long oC, long long iC, float alpha) {
  dim3 g(N / 64, M / 64, batch), b(256);
  k_gemm_bf16<<<g, b, 0, st>>>(A, B, C, res, M, N, K, sAm, sAk, sBk, sBn, sCm,
                               innerB, oA, iA, oB, iB, oC, iC, alpha);
}

extern "C" void kernel_launch(void* const* d_in, const int* in_sizes, int n_in,
                              void* d_out, int out_size, void* d_ws, size_t ws_size,
                              hipStream_t stream) {
  (void)in_sizes; (void)n_in; (void)out_size; (void)ws_size;
  const float* x        = (const float*)d_in[0];
  const float* W_emb    = (const float*)d_in[1];
  const float* b_emb    = (const float*)d_in[2];
  const float* m_ln_w   = (const float*)d_in[3];
  const float* m_Wup    = (const float*)d_in[4];
  const float* m_conv_w = (const float*)d_in[5];
  const float* m_conv_b = (const float*)d_in[6];
  const float* m_Wq     = (const float*)d_in[7];
  const float* m_Wk     = (const float*)d_in[8];
  const float* m_Wv     = (const float*)d_in[9];
  const float* m_Wig    = (const float*)d_in[10];
  const float* m_big    = (const float*)d_in[11];
  const float* m_Wfg    = (const float*)d_in[12];
  const float* m_bfg    = (const float*)d_in[13];
  const float* m_gn_w   = (const float*)d_in[14];
  const float* m_skip   = (const float*)d_in[15];
  const float* m_Wdown  = (const float*)d_in[16];
  const float* s_ln_w   = (const float*)d_in[17];
  const float* s_conv_w = (const float*)d_in[18];
  const float* s_conv_b = (const float*)d_in[19];
  const float* s_Wg     = (const float*)d_in[20];
  const float* s_bg     = (const float*)d_in[21];
  const float* s_R      = (const float*)d_in[22];
  const float* s_gn_w   = (const float*)d_in[23];
  const float* s_ln2_w  = (const float*)d_in[24];
  const float* s_Wff_up = (const float*)d_in[25];
  const float* s_Wff_dn = (const float*)d_in[26];
  const float* post_ln  = (const float*)d_in[27];
  const float* W_out    = (const float*)d_in[28];
  const float* b_out    = (const float*)d_in[29];

  float* ws = (float*)d_ws;
  float* H   = ws + OFF_H;   float* LN  = ws + OFF_LN;  float* UP  = ws + OFF_UP;
  float* XC  = ws + OFF_XC;  float* Q   = ws + OFF_Q;   float* Kb  = ws + OFF_K;
  float* V   = ws + OFF_V;   float* IG  = ws + OFF_IG;  float* FG  = ws + OFF_FG;
  float* CS  = ws + OFF_CS;  float* HAT = ws + OFF_HAT; float* SC  = ws + OFF_SC;
  float* HN  = ws + OFF_HN;  float* GIN = ws + OFF_GIN;
  float* FFU = ws + OFF_FFU; float* FFG = ws + OFF_FFG;
  float* G0  = ws + OFF_G0;                 float* G1 = G0 + (long long)NTOK * 128;
  float* G2  = G1 + (long long)NTOK * 128;  float* G3 = G2 + (long long)NTOK * 128;
  float* YS  = ws + OFF_YS;

  k_embed<<<(NTOK * 128) / 256, 256, 0, stream>>>(x, W_emb, b_emb, H);

  const char bt[6] = {'m', 's', 'm', 's', 'm', 'm'};
  int mi = 0, si = 0;
  for (int blk = 0; blk < 6; ++blk) {
    if (bt[blk] == 'm') {
      k_ln<<<NTOK / 4, 128, 0, stream>>>(H, LN, m_ln_w + mi * 128);
      gemm(stream, LN, m_Wup + (long long)mi * 128 * 512, UP, nullptr,
           NTOK, 512, 128, 128, 1, 512, 1, 512, 1, 1, 0, 0, 0, 0, 0, 0, 1.f);
      k_conv_silu<<<(NTOK * 256) / 256, 256, 0, stream>>>(
          UP, 512, 256, m_conv_w + mi * 1024, m_conv_b + mi * 256, XC);
      k_headwise<<<(NTOK * 64) / 256, 256, 0, stream>>>(
          XC, UP, 512, m_Wq + mi * 1024, m_Wk + mi * 1024, m_Wv + mi * 1024, Q, Kb, V);
      k_gates<<<(NTOK * 4) / 256, 256, 0, stream>>>(
          Q, Kb, V, m_Wig + mi * 3072, m_big + mi * 4,
          m_Wfg + mi * 3072, m_bfg + mi * 4, IG, FG);
      k_cumsum_logsig<<<256, 256, 0, stream>>>(FG, CS);
      // scores[b,h] = q @ k^T * DHM^-0.5   (M=N=256, K=64, batch = 64*4)
      gemm(stream, Q, Kb, SC, nullptr, 256, 256, 64,
           256, 1, 1, 256, 256, 256, 4,
           65536, 64, 65536, 64, 262144, 65536, 0.125f);
      k_rownorm<<<65536, 256, 0, stream>>>(SC, CS, IG);
      // h_attn[b,h] = C~ @ v  (M=256, N=64, K=256)
      gemm(stream, SC, V, HAT, nullptr, 256, 64, 256,
           256, 1, 256, 1, 256, 256, 4,
           262144, 65536, 65536, 64, 65536, 64, 1.f);
      k_mhnorm<<<16384, 128, 0, stream>>>(HAT, HN, m_gn_w + mi * 256, nullptr, 4, 64);
      k_mtail<<<(NTOK * 256) / 256, 256, 0, stream>>>(HN, XC, UP, m_skip + mi * 256, GIN);
      gemm(stream, GIN, m_Wdown + (long long)mi * 256 * 128, H, H,
           NTOK, 128, 256, 256, 1, 128, 1, 128, 1, 1, 0, 0, 0, 0, 0, 0, 1.f);
      ++mi;
    } else {
      k_ln<<<NTOK / 4, 128, 0, stream>>>(H, LN, s_ln_w + si * 128);
      k_conv_silu<<<(NTOK * 128) / 256, 256, 0, stream>>>(
          LN, 128, 128, s_conv_w + si * 512, s_conv_b + si * 128, XC);
      k_sgates<<<(NTOK * 128) / 256, 256, 0, stream>>>(
          XC, LN, s_Wg + si * 16384, s_bg + si * 512, G0, G1, G2, G3);
      k_scan<<<256, 32, 0, stream>>>(G0, G1, G2, G3, s_R + si * 16384, YS);
      k_mhnorm<<<16384, 128, 0, stream>>>(YS, H, s_gn_w + si * 128, H, 4, 32);
      // FFN
      k_ln<<<NTOK / 4, 128, 0, stream>>>(H, LN, s_ln2_w + si * 128);
      gemm(stream, LN, s_Wff_up + (long long)si * 128 * 384, FFU, nullptr,
           NTOK, 384, 128, 128, 1, 384, 1, 384, 1, 1, 0, 0, 0, 0, 0, 0, 1.f);
      k_geglu<<<(NTOK * 192) / 256, 256, 0, stream>>>(FFU, FFG);
      gemm(stream, FFG, s_Wff_dn + (long long)si * 192 * 128, H, H,
           NTOK, 128, 192, 192, 1, 128, 1, 128, 1, 1, 0, 0, 0, 0, 0, 0, 1.f);
      ++si;
    }
  }

  k_final<<<16, 128, 0, stream>>>(H, post_ln, W_out, b_out, (float*)d_out);
}
